// RNN_34359739202
// MI455X (gfx1250) — compile-verified, matrix operand order checked by reading, not run on hardware
//
#include <hip/hip_runtime.h>
#include <hip/hip_bf16.h>

#define T_  2048
#define B_  64
#define E_  200
#define H_  32
#define G_  128   // 4*H
#define O_  2
#define KT_ 7     // ceil(200/32) k-tiles for 16x16x32 WMMA

typedef __attribute__((ext_vector_type(16))) __bf16 v16bf;
typedef __attribute__((ext_vector_type(8)))  float  v8f;
typedef __attribute__((ext_vector_type(2)))  float  f2;   // VGPR pair for v_pk_fma_f32

__device__ __forceinline__ float fast_sigmoid(float x) {
    // 1/(1+2^(-x*log2e)) : v_exp_f32 is exp2 on AMD
    return __builtin_amdgcn_rcpf(1.0f + __builtin_amdgcn_exp2f(-1.4426950408889634f * x));
}
__device__ __forceinline__ float fast_tanh(float x) {
    // tanh = 1 - 2/(exp(2x)+1)
    return 1.0f - 2.0f * __builtin_amdgcn_rcpf(1.0f + __builtin_amdgcn_exp2f(2.8853900817779268f * x));
}

// acc += w * broadcast(h.lo)  (one v_pk_fma_f32, src1 low half replicated)
__device__ __forceinline__ void pk_fma_blo(f2& acc, f2 w, f2 h) {
    asm("v_pk_fma_f32 %0, %1, %2, %0 op_sel:[0,0,0] op_sel_hi:[1,0,1]"
        : "+v"(acc) : "v"(w), "v"(h));
}
// acc += w * broadcast(h.hi)  (one v_pk_fma_f32, src1 high half replicated)
__device__ __forceinline__ void pk_fma_bhi(f2& acc, f2 w, f2 h) {
    asm("v_pk_fma_f32 %0, %1, %2, %0 op_sel:[0,1,0] op_sel_hi:[1,1,1]"
        : "+v"(acc) : "v"(w), "v"(h));
}

// ---------------------------------------------------------------------------
// Kernel 1: embedding gather + input GEMM via WMMA bf16.
//   pre[m, n] = sum_k emb[x[m], k] * W_ih[n, k] + (b_ih[n] + b_hh[n])
// Stored GATE-INTERLEAVED for the scan kernel:
//   xg[m][ (n&31)*4 + (n>>5) ] = pre[m, n]
// so lane j of the scan reads its 4 gate pre-activations as one float4.
// ---------------------------------------------------------------------------
__global__ __launch_bounds__(256) void k1_embed_gemm(
    const int* __restrict__ x, const float* __restrict__ emb,
    const float* __restrict__ W_ih, const float* __restrict__ b_ih,
    const float* __restrict__ b_hh, float* __restrict__ xg)
{
    // W_ih^T staged in LDS in exact B-fragment order: [kt][nt][lane][16 bf16]
    __shared__ __align__(32) __bf16 Bfrag[KT_ * 8 * 32 * 16]; // 57344 B
    __shared__ float biasS[G_];

    for (int f = threadIdx.x; f < KT_ * 8 * 32 * 16; f += 256) {
        int idx = f & 15;
        int ln  = (f >> 4) & 31;
        int nt  = (f >> 9) & 7;
        int kt  = f >> 12;
        int n = nt * 16 + (ln & 15);
        int k = kt * 32 + ((idx & 8) << 1) + ((ln >> 4) << 3) + (idx & 7);
        float v = (k < E_) ? W_ih[(size_t)n * E_ + k] : 0.0f;
        Bfrag[f] = (__bf16)v;
    }
    if (threadIdx.x < G_) biasS[threadIdx.x] = b_ih[threadIdx.x] + b_hh[threadIdx.x];
    __syncthreads();

    const int lane = threadIdx.x & 31;
    const int wave = threadIdx.x >> 5;
    const int half = lane >> 4;          // K-group selector
    const size_t mbase = ((size_t)blockIdx.x * 8 + wave) * 16;
    const size_t m = mbase + (lane & 15);
    const float* erow = emb + (size_t)x[m] * E_;   // gathered embedding row

    v8f acc[8];
    v8f zero = {};
#pragma unroll
    for (int i = 0; i < 8; ++i) acc[i] = zero;

    // A-fragment double buffer: 4 float4 chunks = 16 floats of this lane's row.
    float4 fb0, fb1, fb2, fb3;
    {
        int kb0 = 0 * 32 + half * 8;          // run 0
        int kb1 = 0 * 32 + 16 + half * 8;     // run 1
        fb0 = *(const float4*)(erow + kb0);
        fb1 = *(const float4*)(erow + kb0 + 4);
        fb2 = *(const float4*)(erow + kb1);
        fb3 = *(const float4*)(erow + kb1 + 4);
    }
    float ftail[16];

#pragma unroll
    for (int kt = 0; kt < KT_; ++kt) {
        v16bf a;
        if (kt < 6) {
            a[0]=(__bf16)fb0.x; a[1]=(__bf16)fb0.y; a[2]=(__bf16)fb0.z; a[3]=(__bf16)fb0.w;
            a[4]=(__bf16)fb1.x; a[5]=(__bf16)fb1.y; a[6]=(__bf16)fb1.z; a[7]=(__bf16)fb1.w;
            a[8]=(__bf16)fb2.x; a[9]=(__bf16)fb2.y; a[10]=(__bf16)fb2.z; a[11]=(__bf16)fb2.w;
            a[12]=(__bf16)fb3.x; a[13]=(__bf16)fb3.y; a[14]=(__bf16)fb3.z; a[15]=(__bf16)fb3.w;
        } else {
#pragma unroll
            for (int u = 0; u < 16; ++u) a[u] = (__bf16)ftail[u];
        }

        // Issue next k-tile's A loads so they overlap this tile's 8 WMMAs.
        if (kt < 5) {
            int kb0 = (kt + 1) * 32 + half * 8;
            int kb1 = (kt + 1) * 32 + 16 + half * 8;
            fb0 = *(const float4*)(erow + kb0);
            fb1 = *(const float4*)(erow + kb0 + 4);
            fb2 = *(const float4*)(erow + kb1);
            fb3 = *(const float4*)(erow + kb1 + 4);
        } else if (kt == 5) {
            // tail tile (k = 192..223): element-wise guarded, zero-padded
#pragma unroll
            for (int run = 0; run < 2; ++run) {
#pragma unroll
                for (int u = 0; u < 8; ++u) {
                    int k = 6 * 32 + run * 16 + half * 8 + u;
                    ftail[run * 8 + u] = (k < E_) ? erow[k] : 0.0f;
                }
            }
        }

#pragma unroll
        for (int nt = 0; nt < 8; ++nt) {
            v16bf bf = *(const v16bf*)&Bfrag[(((kt * 8) + nt) * 32 + lane) * 16];
            acc[nt] = __builtin_amdgcn_wmma_f32_16x16x32_bf16(
                false, a, false, bf, (short)0, acc[nt], false, false);
        }
    }

    // C layout: element v -> row mbase + v + 8*half, col nt*16 + (lane&15)
#pragma unroll
    for (int nt = 0; nt < 8; ++nt) {
        int n = nt * 16 + (lane & 15);
        int np = (n & 31) * 4 + (n >> 5);   // gate-interleaved position
        float bv = biasS[n];
#pragma unroll
        for (int v = 0; v < 8; ++v) {
            size_t mm = mbase + v + 8 * half;
            xg[mm * G_ + np] = acc[nt][v] + bv;
        }
    }
}

// ---------------------------------------------------------------------------
// Kernel 2: sequential LSTM scan. One SINGLE-WAVE block per batch row
// (whole SIMD + VGPR file per row; barriers are free). Lane j owns h[j],
// c[j] and W_hh rows {j,32+j,64+j,96+j} packed as float2 (i,f)/(g,o) pairs.
// 64 v_pk_fma_f32 per step (forced via inline asm, src1 op_sel broadcast),
// split over 8 independent accumulator chains. One float4 xg load per step
// (gate-interleaved), software-pipelined one step ahead + deep prefetch.
// ---------------------------------------------------------------------------
__global__ __launch_bounds__(32) void k2_lstm_scan(
    const float* __restrict__ xg, const float* __restrict__ W_hh,
    float* __restrict__ hs)
{
    const int lane = threadIdx.x;     // 0..31
    const int b = blockIdx.x;         // 0..63

    __shared__ float hbuf[H_];

    // Recurrent weights in VGPRs, packed per gate-pair.
    f2 Wif[H_], Wgo[H_];
#pragma unroll
    for (int k = 0; k < H_; k += 4) {
        float4 a0 = *(const float4*)(W_hh + (size_t)(       lane) * H_ + k);
        float4 a1 = *(const float4*)(W_hh + (size_t)(32 +   lane) * H_ + k);
        float4 a2 = *(const float4*)(W_hh + (size_t)(64 +   lane) * H_ + k);
        float4 a3 = *(const float4*)(W_hh + (size_t)(96 +   lane) * H_ + k);
        Wif[k+0] = (f2){a0.x, a1.x}; Wgo[k+0] = (f2){a2.x, a3.x};
        Wif[k+1] = (f2){a0.y, a1.y}; Wgo[k+1] = (f2){a2.y, a3.y};
        Wif[k+2] = (f2){a0.z, a1.z}; Wgo[k+2] = (f2){a2.z, a3.z};
        Wif[k+3] = (f2){a0.w, a1.w}; Wgo[k+3] = (f2){a2.w, a3.w};
    }

    float c = 0.0f;
    hbuf[lane] = 0.0f;
    __syncthreads();   // single-wave workgroup: lowers to S_NOP

    const float* xgb = xg + (size_t)b * G_;
    // software pipeline: xv holds step t's gate pre-activations
    float4 xv = *(const float4*)(xgb + (size_t)lane * 4);

    for (int t = 0; t < T_; ++t) {
        // issue next step's load + deep prefetch first (off critical path)
        int tn = (t + 1 < T_) ? (t + 1) : t;
        float4 xnext = *(const float4*)(xgb + (size_t)tn * (B_ * G_) + lane * 4);
        if (t + 13 < T_)
            __builtin_prefetch(xgb + (size_t)(t + 13) * (B_ * G_) + lane * 4, 0, 0);

        f2 z = {};
        f2 rif0 = (f2){xv.x, xv.y}, rif1 = z, rif2 = z, rif3 = z;
        f2 rgo0 = (f2){xv.z, xv.w}, rgo1 = z, rgo2 = z, rgo3 = z;

#pragma unroll
        for (int k = 0; k < H_; k += 4) {
            float4 hv = *(const float4*)&hbuf[k];   // broadcast ds_load_b128
            f2 hlo; hlo.x = hv.x; hlo.y = hv.y;     // aliases hv's VGPR pair
            f2 hhi; hhi.x = hv.z; hhi.y = hv.w;
            pk_fma_blo(rif0, Wif[k+0], hlo);  pk_fma_blo(rgo0, Wgo[k+0], hlo);
            pk_fma_bhi(rif1, Wif[k+1], hlo);  pk_fma_bhi(rgo1, Wgo[k+1], hlo);
            pk_fma_blo(rif2, Wif[k+2], hhi);  pk_fma_blo(rgo2, Wgo[k+2], hhi);
            pk_fma_bhi(rif3, Wif[k+3], hhi);  pk_fma_bhi(rgo3, Wgo[k+3], hhi);
        }
        f2 rif = (rif0 + rif1) + (rif2 + rif3);
        f2 rgo = (rgo0 + rgo1) + (rgo2 + rgo3);

        float iv = fast_sigmoid(rif.x);
        float fv = fast_sigmoid(rif.y);
        float gv = fast_tanh(rgo.x);
        float ov = fast_sigmoid(rgo.y);
        c = fv * c + iv * gv;
        float h = ov * fast_tanh(c);

        hbuf[lane] = h;                  // visible to own wave next step (lockstep)
        __builtin_amdgcn_wave_barrier(); // keep compiler from reordering across it
        hs[((size_t)t * B_ + b) * H_ + lane] = h;
        xv = xnext;
    }
}

// ---------------------------------------------------------------------------
// Kernel 3: logits = hs @ W_out^T + b_out, then log_softmax over the BATCH
// axis (axis=1 of [T,B,O]). One 64-thread block per t.
// ---------------------------------------------------------------------------
__global__ __launch_bounds__(64) void k3_logits_lsm(
    const float* __restrict__ hs, const float* __restrict__ W_out,
    const float* __restrict__ b_out, float* __restrict__ out)
{
    const int t = blockIdx.x;
    const int b = threadIdx.x;   // 0..63
    __shared__ float L0[B_], L1[B_];

    const float* hp = hs + ((size_t)t * B_ + b) * H_;
    float l0 = b_out[0], l1 = b_out[1];
#pragma unroll
    for (int k = 0; k < H_; k += 4) {
        float4 hv = *(const float4*)(hp + k);
        float4 w0 = *(const float4*)(W_out + k);
        float4 w1 = *(const float4*)(W_out + H_ + k);
        l0 += hv.x*w0.x + hv.y*w0.y + hv.z*w0.z + hv.w*w0.w;
        l1 += hv.x*w1.x + hv.y*w1.y + hv.z*w1.z + hv.w*w1.w;
    }
    L0[b] = l0; L1[b] = l1;
    __syncthreads();

    float m0 = -3.402823466e38f, m1 = m0;
    for (int i = 0; i < B_; ++i) { m0 = fmaxf(m0, L0[i]); m1 = fmaxf(m1, L1[i]); }
    float s0 = 0.0f, s1 = 0.0f;
    for (int i = 0; i < B_; ++i) {
        s0 += __builtin_amdgcn_exp2f(1.4426950408889634f * (L0[i] - m0));
        s1 += __builtin_amdgcn_exp2f(1.4426950408889634f * (L1[i] - m1));
    }
    float ls0 = m0 + 0.69314718055994531f * __builtin_amdgcn_logf(s0);
    float ls1 = m1 + 0.69314718055994531f * __builtin_amdgcn_logf(s1);

    size_t o = ((size_t)t * B_ + b) * O_;
    out[o + 0] = l0 - ls0;
    out[o + 1] = l1 - ls1;
}

// ---------------------------------------------------------------------------
extern "C" void kernel_launch(void* const* d_in, const int* in_sizes, int n_in,
                              void* d_out, int out_size, void* d_ws, size_t ws_size,
                              hipStream_t stream) {
    const int*   x     = (const int*)  d_in[0];
    const float* emb   = (const float*)d_in[1];
    const float* W_ih  = (const float*)d_in[2];
    const float* W_hh  = (const float*)d_in[3];
    const float* b_ih  = (const float*)d_in[4];
    const float* b_hh  = (const float*)d_in[5];
    const float* W_out = (const float*)d_in[6];
    const float* b_out = (const float*)d_in[7];
    float* out = (float*)d_out;

    float* xg = (float*)d_ws;                      // T*B*128 f32 = 67 MB
    float* hs = xg + (size_t)T_ * B_ * G_;         // T*B*32  f32 = 16.8 MB

    // 8192 M-tiles of 16 rows, 8 waves (tiles) per block
    k1_embed_gemm<<<1024, 256, 0, stream>>>(x, emb, W_ih, b_ih, b_hh, xg);
    // 64 batch rows, one single-wave block each (whole SIMD per row)
    k2_lstm_scan<<<64, 32, 0, stream>>>(xg, W_hh, hs);
    // one block per timestep; softmax over batch axis
    k3_logits_lsm<<<T_, 64, 0, stream>>>(hs, W_out, b_out, out);
}
